// General_SEBlock_16801912062782
// MI455X (gfx1250) — compile-verified, matrix operand order checked by reading, not run on hardware
//
#include <hip/hip_runtime.h>
#include <hip/hip_bf16.h>

// ---------------------------------------------------------------------------
// Types for CDNA5 WMMA (wave32)
// ---------------------------------------------------------------------------
typedef __attribute__((ext_vector_type(16))) __bf16 v16bf;
typedef __attribute__((ext_vector_type(8)))  __bf16 v8bf;
typedef __attribute__((ext_vector_type(8)))  float  v8f;

#define K_DIM      256
#define NPIX       4096        // 64*64 per (b,c)
#define NBATCH     32
#define PIX_BLK    64          // pixels per workgroup in gemm
#define KSTEP      32          // wmma bf16 K per instruction
#define NKSTEPS    (K_DIM / KSTEP)

// round-to-nearest-even float -> bf16 (scalar, used in mlp2 only)
__device__ __forceinline__ __bf16 f2bf(float f) {
    union { float f; unsigned u; } v; v.f = f;
    unsigned r = (v.u + 0x7FFFu + ((v.u >> 16) & 1u)) >> 16;
    unsigned short s = (unsigned short)r;
    __bf16 out;
    __builtin_memcpy(&out, &s, 2);
    return out;
}

// pack two floats into one dword of 2x bf16 (lo = a, hi = b)
// round-to-nearest (ties away): tie-break bias is far below bf16 quantization.
__device__ __forceinline__ unsigned pack_bf16x2(float a, float b) {
    union { float f; unsigned u; } ua, ub; ua.f = a; ub.f = b;
    return ((ua.u + 0x8000u) >> 16) | ((ub.u + 0x8000u) & 0xFFFF0000u);
}

// ---------------------------------------------------------------------------
// Kernel 1: y[c] = mean over (b,h,w) of x[b,c,h,w]
// grid = 256 (one block per channel), block = 256
// ---------------------------------------------------------------------------
__global__ void se_mean_kernel(const float* __restrict__ x, float* __restrict__ y) {
    const int c = blockIdx.x;
    const int t = threadIdx.x;
    __shared__ float red[256];

    float sum = 0.0f;
    for (int b = 0; b < NBATCH; ++b) {
        const float4* base = (const float4*)(x + ((size_t)b * K_DIM + c) * NPIX);
        #pragma unroll
        for (int j = 0; j < 4; ++j) {
            float4 v = base[t + j * 256];
            sum += v.x + v.y + v.z + v.w;
        }
    }
    red[t] = sum;
    __syncthreads();
    for (int s = 128; s > 0; s >>= 1) {
        if (t < s) red[t] += red[t + s];
        __syncthreads();
    }
    if (t == 0) y[c] = red[0] * (1.0f / (float)(NBATCH * NPIX));
}

// ---------------------------------------------------------------------------
// Kernel 2: h[i,o] = relu( dot(W1[i,o,:], y) + b1[i,o] )
// grid = 257, block = 256 (one thread per o)
// ---------------------------------------------------------------------------
__global__ void se_mlp1_kernel(const float* __restrict__ W1, const float* __restrict__ b1,
                               const float* __restrict__ y, float* __restrict__ h) {
    const int i = blockIdx.x;
    const int t = threadIdx.x;
    __shared__ float ys[K_DIM];
    ys[t] = y[t];
    __syncthreads();

    const float4* w = (const float4*)(W1 + ((size_t)i * K_DIM + t) * K_DIM);
    float acc = b1[(size_t)i * K_DIM + t];
    #pragma unroll 8
    for (int j = 0; j < K_DIM / 4; ++j) {
        float4 wv = w[j];
        acc += wv.x * ys[4 * j + 0] + wv.y * ys[4 * j + 1] +
               wv.z * ys[4 * j + 2] + wv.w * ys[4 * j + 3];
    }
    h[(size_t)i * K_DIM + t] = fmaxf(acc, 0.0f);
}

// ---------------------------------------------------------------------------
// Kernel 3: kern[i,o] = dot(W2[i,o,:], h[i,:]) + b2[i,o]
//   i<256  -> kern_bf[i*256+o]  (bf16, A-matrix row-major [o_conv][c])
//   i==256 -> bias[o]           (fp32)
// grid = 257, block = 256
// ---------------------------------------------------------------------------
__global__ void se_mlp2_kernel(const float* __restrict__ W2, const float* __restrict__ b2,
                               const float* __restrict__ h, __bf16* __restrict__ kern_bf,
                               float* __restrict__ bias) {
    const int i = blockIdx.x;
    const int t = threadIdx.x;
    __shared__ float hs[K_DIM];
    hs[t] = h[(size_t)i * K_DIM + t];
    __syncthreads();

    const float4* w = (const float4*)(W2 + ((size_t)i * K_DIM + t) * K_DIM);
    float acc = b2[(size_t)i * K_DIM + t];
    #pragma unroll 8
    for (int j = 0; j < K_DIM / 4; ++j) {
        float4 wv = w[j];
        acc += wv.x * hs[4 * j + 0] + wv.y * hs[4 * j + 1] +
               wv.z * hs[4 * j + 2] + wv.w * hs[4 * j + 3];
    }
    if (i < 256) kern_bf[(size_t)i * K_DIM + t] = f2bf(acc);
    else         bias[t] = acc;
}

// ---------------------------------------------------------------------------
// Fragment load from LDS: rows of 32 bf16; lane picks row, two 16B K-chunks.
// ISA 16-bit A/B layout: lanes 0-15 hold K=[k..k+7] and [k+16..k+23];
// lanes 16-31 hold K=[k+8..k+15] and [k+24..k+31].
// ---------------------------------------------------------------------------
__device__ __forceinline__ v16bf load_frag(const __bf16* lds, int row, int half) {
    const __bf16* p = lds + row * KSTEP + half * 8;
    v8bf c0 = *(const v8bf*)(p);
    v8bf c1 = *(const v8bf*)(p + 16);
    v16bf f;
    #pragma unroll
    for (int e = 0; e < 8; ++e) { f[e] = c0[e]; f[e + 8] = c1[e]; }
    return f;
}

// ---------------------------------------------------------------------------
// Kernel 4: out[b,o,p] = sum_c kern[o,c] * x[b,c,p] + bias[o]
// grid = (64 pixel-blocks, 32 batches), block = 256 (8 waves)
// Double-buffered LDS (one barrier per K-step); each wave:
// M-tiles {2w, 2w+1} x N-tiles {0..3} => 8 wmma per K-step.
// ---------------------------------------------------------------------------
__global__ void __launch_bounds__(256)
se_gemm_kernel(const float* __restrict__ x, const __bf16* __restrict__ kern_bf,
               const float* __restrict__ bias, float* __restrict__ out) {
    const int b  = blockIdx.y;
    const int p0 = blockIdx.x * PIX_BLK;
    const int t  = threadIdx.x;
    const int wv = t >> 5;         // wave id 0..7
    const int ln = t & 31;         // lane  0..31
    const int half = ln >> 4;      // 0 or 1
    const int col  = ln & 15;

    // double-buffered tiles: A = kern [M=256][K=32], B = x^T [N=64][K=32]
    __shared__ __attribute__((aligned(16))) __bf16 sA[2][256 * KSTEP];     // 2 x 16 KB
    __shared__ __attribute__((aligned(16))) __bf16 sB[2][PIX_BLK * KSTEP]; // 2 x  4 KB

    // staging helpers -------------------------------------------------------
    auto stageA = [&](int k0, __bf16* dst) {
        const v8bf* g = (const v8bf*)(kern_bf + (size_t)t * K_DIM + k0);
        v8bf* s = (v8bf*)(dst + t * KSTEP);
        #pragma unroll
        for (int j = 0; j < 4; ++j) s[j] = g[j];      // 4x ds_store_b128
    };
    auto stageB = [&](int k0, __bf16* dst) {
        // thread t: K-pair kp..kp+1, 4 consecutive pixels pl..pl+3
        const int kp = (t & 15) * 2;                  // 0,2,...,30
        const int pl = (t >> 4) * 4;                  // 0,4,...,60
        const float* g0 = x + ((size_t)b * K_DIM + (k0 + kp)) * NPIX + p0 + pl;
        const float* g1 = g0 + NPIX;
        float4 va = *(const float4*)g0;               // channel kp
        float4 vb = *(const float4*)g1;               // channel kp+1
        unsigned* d = (unsigned*)dst;                 // [pixel][K] as dword pairs
        const int kd = kp >> 1;
        d[(pl + 0) * (KSTEP / 2) + kd] = pack_bf16x2(va.x, vb.x);
        d[(pl + 1) * (KSTEP / 2) + kd] = pack_bf16x2(va.y, vb.y);
        d[(pl + 2) * (KSTEP / 2) + kd] = pack_bf16x2(va.z, vb.z);
        d[(pl + 3) * (KSTEP / 2) + kd] = pack_bf16x2(va.w, vb.w);
        // prefetch next-next K-step of x while WMMAs run (near-cache locality)
        if (k0 + KSTEP < K_DIM)
            __builtin_prefetch(g0 + (size_t)KSTEP * NPIX, 0, 3);
    };

    v8f acc[2][4] = {};

    // prologue: fill buffer 0
    stageA(0, sA[0]);
    stageB(0, sB[0]);

    for (int k = 0; k < NKSTEPS; ++k) {
        __syncthreads();                 // buf(k&1) visible; buf(k&1^1) free
        const int cur = k & 1;
        if (k + 1 < NKSTEPS) {           // stage next tile into the other buffer
            stageA((k + 1) * KSTEP, sA[cur ^ 1]);
            stageB((k + 1) * KSTEP, sB[cur ^ 1]);
        }

        // fragments + WMMA from current buffer
        v16bf af[2], bf[4];
        #pragma unroll
        for (int i = 0; i < 2; ++i)
            af[i] = load_frag(sA[cur], (2 * wv + i) * 16 + col, half);
        #pragma unroll
        for (int n = 0; n < 4; ++n)
            bf[n] = load_frag(sB[cur], n * 16 + col, half);

        #pragma unroll
        for (int i = 0; i < 2; ++i)
            #pragma unroll
            for (int n = 0; n < 4; ++n)
                acc[i][n] = __builtin_amdgcn_wmma_f32_16x16x32_bf16(
                    false, af[i], false, bf[n], (short)0, acc[i][n], false, false);
    }

    // epilogue: C/D layout VGPR r, lane ln: M = mb + r (mb already holds half*8), N = col
    #pragma unroll
    for (int i = 0; i < 2; ++i) {
        const int mb = (2 * wv + i) * 16 + half * 8;
        float4 b0 = *(const float4*)(bias + mb);
        float4 b1 = *(const float4*)(bias + mb + 4);
        float bb[8] = { b0.x, b0.y, b0.z, b0.w, b1.x, b1.y, b1.z, b1.w };
        #pragma unroll
        for (int n = 0; n < 4; ++n) {
            const int p = p0 + n * 16 + col;
            float* op = out + ((size_t)b * K_DIM + mb) * NPIX + p;
            #pragma unroll
            for (int r = 0; r < 8; ++r)
                op[(size_t)r * NPIX] = acc[i][n][r] + bb[r];
        }
    }
}

// ---------------------------------------------------------------------------
// Launcher
// ---------------------------------------------------------------------------
extern "C" void kernel_launch(void* const* d_in, const int* in_sizes, int n_in,
                              void* d_out, int out_size, void* d_ws, size_t ws_size,
                              hipStream_t stream) {
    (void)in_sizes; (void)n_in; (void)out_size; (void)ws_size;
    const float* x  = (const float*)d_in[0];
    const float* W1 = (const float*)d_in[1];
    const float* b1 = (const float*)d_in[2];
    const float* W2 = (const float*)d_in[3];
    const float* b2 = (const float*)d_in[4];
    float* out = (float*)d_out;

    // workspace layout (16B-aligned sections)
    char* ws = (char*)d_ws;
    float*  y       = (float*)(ws + 0);                         //   256 f32
    float*  h       = (float*)(ws + 1024);                      // 257*256 f32
    __bf16* kern_bf = (__bf16*)(ws + 1024 + 257 * 256 * 4);     // 256*256 bf16
    float*  bias    = (float*)(ws + 1024 + 257 * 256 * 4 + 256 * 256 * 2); // 256 f32

    se_mean_kernel<<<256, 256, 0, stream>>>(x, y);
    se_mlp1_kernel<<<257, 256, 0, stream>>>(W1, b1, y, h);
    se_mlp2_kernel<<<257, 256, 0, stream>>>(W2, b2, h, kern_bf, bias);

    dim3 grid(NPIX / PIX_BLK, NBATCH);   // (64, 32)
    se_gemm_kernel<<<grid, 256, 0, stream>>>(x, kern_bf, bias, out);
}
